// Reqo_47124381171918
// MI455X (gfx1250) — matrix operand order repaired and checked by reading, not compiled, not run on hardware
//
#include <hip/hip_runtime.h>

// ---------------------------------------------------------------------------
// MI455X (gfx1250) implementation. All dense GEMMs run through
// v_wmma_f32_16x16x32_bf16 (wave32). Each wave computes a 32x32 output
// macro-tile: 2 A-fragments x 2 B-fragments -> 4 WMMAs per K-step, which
// halves fragment VMEM traffic per FLOP vs a 16x16 tile. fp32 activations
// are converted to bf16 in-register (v_cvt_pk_bf16_f32); weights are
// transposed+converted to bf16 scratch before each GEMM and stay resident
// in L2 (192 MB) across the whole grid. A-rows are streamed from HBM with
// global_prefetch on the next K-block.
// ---------------------------------------------------------------------------

#define N_NODES 65536
#define RAW     688
#define NT_     26
#define TBL_    20
#define NTE_    32
#define NFD_    374
#define NFDP    384
#define EMB_    128
#define GG_     2048   // B*S groups
#define LL_     32     // seq len
#define BB_     128    // batch

typedef __attribute__((ext_vector_type(16))) __bf16 v16bf;
typedef __attribute__((ext_vector_type(8)))  __bf16 v8bf;
typedef __attribute__((ext_vector_type(8)))  float  v8f;
typedef __attribute__((ext_vector_type(4)))  float  v4f;

union BFrag { v16bf v; v8bf h[2]; };

__device__ __forceinline__ float sigf(float x){ return 1.0f/(1.0f+expf(-x)); }

// A-fragment per lane (row = lane&15, half = lane>>4):
//   elems 0..7  -> K = k0 + 8*half .. +7        (two v4f f32 loads, cvt bf16)
//   elems 8..15 -> K = k0 + 16 + 8*half .. +7
__device__ __forceinline__ v16bf load_a_frag(const float* __restrict__ arow, int kA)
{
    v4f f0 = *(const v4f*)(arow + kA);
    v4f f1 = *(const v4f*)(arow + kA + 4);
    v4f f2 = *(const v4f*)(arow + kA + 16);
    v4f f3 = *(const v4f*)(arow + kA + 20);
    v16bf av;
#pragma unroll
    for (int i = 0; i < 4; ++i) {
        av[i]      = (__bf16)f0[i];
        av[i + 4]  = (__bf16)f1[i];
        av[i + 8]  = (__bf16)f2[i];
        av[i + 12] = (__bf16)f3[i];
    }
    return av;
}

// B-fragment per lane (col = lane&15): K = k0 + 8*half .. +7, k0 + 16 + 8*half .. +7 (two b128 loads)
__device__ __forceinline__ v16bf load_b_frag(const __bf16* __restrict__ brow, int kB)
{
    BFrag b;
    b.h[0] = *(const v8bf*)(brow + kB);
    b.h[1] = *(const v8bf*)(brow + kB + 16);
    return b.v;
}

// ---------------------------------------------------------------------------
// WMMA GEMM: C[M x Nout] = act(A[M x K(f32)] * Wt[Nout x Kpad(bf16)]^T + bias)
// One wave per 32x32 macro-tile (4 x v_wmma_f32_16x16x32_bf16 per K-step).
// Kloop multiple of 32; M, Nout multiples of 32.
// C layout per 16x16 sub-tile: col = lane&15, row = 8*(lane>>4) + r.
// ---------------------------------------------------------------------------
__global__ void gemm_bf16_wmma(const float* __restrict__ A, int lda,
                               const __bf16* __restrict__ Wt, int ldw,
                               const float* __restrict__ bias,
                               float* __restrict__ C, int ldc,
                               int M, int Nout, int Kloop, int relu_act)
{
    int gw   = (int)(((long)blockIdx.x * blockDim.x + threadIdx.x) >> 5);
    int lane = threadIdx.x & 31;
    int tilesN = Nout >> 5;
    int tilesM = M >> 5;
    if (gw >= tilesM * tilesN) return;
    int tm = gw / tilesN, tn = gw % tilesN;
    int half = lane >> 4, rc = lane & 15;

    const float*  arow0 = A  + (size_t)(tm * 32 + rc) * lda;
    const float*  arow1 = arow0 + (size_t)16 * lda;
    const __bf16* brow0 = Wt + (size_t)(tn * 32 + rc) * ldw;
    const __bf16* brow1 = brow0 + (size_t)16 * ldw;

    v8f acc00 = {0.f,0.f,0.f,0.f,0.f,0.f,0.f,0.f};
    v8f acc01 = acc00, acc10 = acc00, acc11 = acc00;

    for (int k0 = 0; k0 < Kloop; k0 += 32) {
        int kA = k0 + 8 * half;
        int kB = k0 + 8 * half;
        // prefetch next K-block of the streamed A rows (global_prefetch_b8)
        __builtin_prefetch(arow0 + kA + 32, 0, 1);
        __builtin_prefetch(arow1 + kA + 32, 0, 1);
        v16bf a0 = load_a_frag(arow0, kA);
        v16bf a1 = load_a_frag(arow1, kA);
        v16bf b0 = load_b_frag(brow0, kB);
        v16bf b1 = load_b_frag(brow1, kB);
        acc00 = __builtin_amdgcn_wmma_f32_16x16x32_bf16(false, a0, false, b0, (short)0, acc00, false, false);
        acc01 = __builtin_amdgcn_wmma_f32_16x16x32_bf16(false, a0, false, b1, (short)0, acc01, false, false);
        acc10 = __builtin_amdgcn_wmma_f32_16x16x32_bf16(false, a1, false, b0, (short)0, acc10, false, false);
        acc11 = __builtin_amdgcn_wmma_f32_16x16x32_bf16(false, a1, false, b1, (short)0, acc11, false, false);
    }

    int col0 = tn * 32 + rc;
    int col1 = col0 + 16;
    float bb0 = bias ? bias[col0] : 0.0f;
    float bb1 = bias ? bias[col1] : 0.0f;
    int row0 = tm * 32 + half * 8;
    float* c0 = C + (size_t)row0 * ldc;
    float* c1 = C + (size_t)(row0 + 16) * ldc;
#pragma unroll
    for (int r = 0; r < 8; ++r) {
        float v00 = acc00[r] + bb0;
        float v01 = acc01[r] + bb1;
        float v10 = acc10[r] + bb0;
        float v11 = acc11[r] + bb1;
        if (relu_act) {
            v00 = fmaxf(v00, 0.f); v01 = fmaxf(v01, 0.f);
            v10 = fmaxf(v10, 0.f); v11 = fmaxf(v11, 0.f);
        }
        c0[(size_t)r * ldc + col0] = v00;
        c0[(size_t)r * ldc + col1] = v01;
        c1[(size_t)r * ldc + col0] = v10;
        c1[(size_t)r * ldc + col1] = v11;
    }
}

// W[Kdim x Nout] (row-major) -> Wt[Nout x Kpad] bf16, zero-padded K.
__global__ void transpose_w_k(const float* __restrict__ W, __bf16* __restrict__ Wt,
                              int Kdim, int Nout, int Kpad, long n)
{
    long i = (long)blockIdx.x * blockDim.x + threadIdx.x;
    if (i >= n) return;
    int nn = (int)(i / Kpad), k = (int)(i % Kpad);
    Wt[i] = (k < Kdim) ? (__bf16)W[(size_t)k * Nout + nn] : (__bf16)0.0f;
}

// GRU weights are already [out x in] -> plain convert.
__global__ void convert_w_k(const float* __restrict__ W, __bf16* __restrict__ Wt, long n)
{
    long i = (long)blockIdx.x * blockDim.x + threadIdx.x;
    if (i >= n) return;
    Wt[i] = (__bf16)W[i];
}

__global__ void fill_f32(float* p, float v, long n)
{
    long i = (long)blockIdx.x * blockDim.x + threadIdx.x;
    if (i < n) p[i] = v;
}

// ------------------------- encoder ----------------------------------------
__global__ void copy_stats_tbl(const float* __restrict__ nf, float* __restrict__ xf, long n)
{
    long i = (long)blockIdx.x * blockDim.x + threadIdx.x;
    if (i >= n) return;
    int node = (int)(i / 22), j = (int)(i % 22);
    xf[(size_t)node * NFDP + NTE_ + j] = nf[(size_t)node * RAW + NT_ + j];
}

__global__ void colenc_k(const float* __restrict__ nf, const float* __restrict__ Wcol,
                         float* __restrict__ xf, long n)
{
    long i = (long)blockIdx.x * blockDim.x + threadIdx.x;
    if (i >= n) return;
    int node = (int)(i / 320), r = (int)(i % 320);
    int t = r >> 4, e = r & 15;
    const float* b = nf + (size_t)node * RAW + (NT_ + 2 + TBL_);
    float m = 0.f;
    for (int cc = 0; cc < 4; ++cc) {
        int col = t * 4 + cc;
        float s = 0.f;
#pragma unroll
        for (int f = 0; f < 8; ++f)
            s += b[col * 8 + f] * Wcol[(size_t)(col * 8 + f) * 16 + e];
        s = fmaxf(s, 0.f);
        m = (cc == 0) ? s : fmaxf(m, s);
    }
    xf[(size_t)node * NFDP + (NTE_ + 2 + TBL_) + r] = m;
}

__global__ void padx_k(float* __restrict__ xf, long n)
{
    long i = (long)blockIdx.x * blockDim.x + threadIdx.x;
    if (i >= n) return;
    int node = (int)(i / 10), j = (int)(i % 10);
    xf[(size_t)node * NFDP + NFD_ + j] = 0.f;
}

// ------------------------- edges (transformer conv) ------------------------
__global__ void edge_logits_k(const float* __restrict__ Q, const float* __restrict__ K,
                              const int* __restrict__ src, const int* __restrict__ dst,
                              float* __restrict__ logits, int E)
{
    long idx = (long)blockIdx.x * blockDim.x + threadIdx.x;
    int lane = threadIdx.x & 31;
    long wv = idx >> 5;
    long e = wv >> 1; int h = (int)(wv & 1);
    if (e >= E) return;
    int s = src[e], d = dst[e];
    const float* q = Q + (size_t)d * 256 + h * 128;
    const float* k = K + (size_t)s * 256 + h * 128;
    float sum = 0.f;
#pragma unroll
    for (int c = 0; c < 4; ++c) sum += q[lane + 32 * c] * k[lane + 32 * c];
#pragma unroll
    for (int o = 16; o > 0; o >>= 1) sum += __shfl_xor(sum, o, 32);
    if (lane == 0) logits[e * 2 + h] = sum * 0.088388347648318447f; // 1/sqrt(128)
}

__device__ __forceinline__ void atomicMaxF(float* addr, float val)
{
    if (val >= 0.f) atomicMax((int*)addr, __float_as_int(val));
    else            atomicMin((unsigned int*)addr, __float_as_uint(val));
}

__global__ void seg_max_k(const float* __restrict__ logits, const int* __restrict__ dst,
                          float* __restrict__ smax, long n)
{
    long i = (long)blockIdx.x * blockDim.x + threadIdx.x;
    if (i >= n) return;
    long e = i >> 1; int h = (int)(i & 1);
    atomicMaxF(&smax[(size_t)dst[e] * 2 + h], logits[i]);
}

__global__ void seg_expsum_k(float* __restrict__ logits, const int* __restrict__ dst,
                             const float* __restrict__ smax, float* __restrict__ ssum, long n)
{
    long i = (long)blockIdx.x * blockDim.x + threadIdx.x;
    if (i >= n) return;
    long e = i >> 1; int h = (int)(i & 1);
    int d = dst[e];
    float ex = expf(logits[i] - smax[(size_t)d * 2 + h]);
    logits[i] = ex;
    atomicAdd(&ssum[(size_t)d * 2 + h], ex);
}

// out[d] += 0.5 * a * v[s]  (0.5 folds the head-mean into the scatter)
__global__ void edge_agg_k(const float* __restrict__ alpha, const float* __restrict__ V,
                           const int* __restrict__ src, const int* __restrict__ dst,
                           const float* __restrict__ ssum, float* __restrict__ agg, int E)
{
    long idx = (long)blockIdx.x * blockDim.x + threadIdx.x;
    int lane = threadIdx.x & 31;
    long wv = idx >> 5;
    long e = wv >> 1; int h = (int)(wv & 1);
    if (e >= E) return;
    int s = src[e], d = dst[e];
    float a = 0.5f * alpha[e * 2 + h] / (ssum[(size_t)d * 2 + h] + 1e-16f);
    const float* v = V + (size_t)s * 256 + h * 128;
    float* o = agg + (size_t)d * 128;
#pragma unroll
    for (int c = 0; c < 4; ++c) atomicAdd(&o[lane + 32 * c], a * v[lane + 32 * c]);
}

// ------------------------- combine + batchnorm -----------------------------
__global__ void combine_k(const float* __restrict__ aggi, const float* __restrict__ aggo,
                          float* __restrict__ xlin, float* __restrict__ bns,
                          float* __restrict__ bnq, long n)
{
    long i = (long)blockIdx.x * blockDim.x + threadIdx.x;
    if (i >= n) return;
    int c = (int)(i & 127);
    float v = 0.5f * aggo[i] + 0.5f * aggi[i] + xlin[i]; // ALPHA = 0.5
    xlin[i] = v;
    atomicAdd(&bns[c], v);
    atomicAdd(&bnq[c], v * v);
}

__global__ void bn_final_k(const float* bns, const float* bnq, float* mu, float* iv)
{
    int c = threadIdx.x;
    float m = bns[c] * (1.0f / N_NODES);
    float var = bnq[c] * (1.0f / N_NODES) - m * m;
    mu[c] = m;
    iv[c] = rsqrtf(var + 1e-5f);
}

__global__ void bn_apply_k(const float* __restrict__ xlin, const float* __restrict__ mu,
                           const float* __restrict__ iv, const float* __restrict__ g,
                           const float* __restrict__ b, float* __restrict__ xf, long n)
{
    long i = (long)blockIdx.x * blockDim.x + threadIdx.x;
    if (i >= n) return;
    int node = (int)(i >> 7), c = (int)(i & 127);
    float v = (xlin[i] - mu[c]) * iv[c] * g[c] + b[c];
    xf[(size_t)node * NFDP + c] = fmaxf(v, 0.f);
}

// ------------------------- GRU ---------------------------------------------
__global__ void gru_reorder_k(const float* __restrict__ xf, float* __restrict__ xseq, long n)
{
    long i = (long)blockIdx.x * blockDim.x + threadIdx.x;
    if (i >= n) return;
    int node = (int)(i >> 7), c = (int)(i & 127);
    int g = node >> 5, t = node & 31;
    xseq[((size_t)t * GG_ + g) * EMB_ + c] = xf[(size_t)node * NFDP + c];
}

__global__ void gru_update_k(const float* __restrict__ gi, const float* __restrict__ gh,
                             float* __restrict__ h, long n)
{
    long i = (long)blockIdx.x * blockDim.x + threadIdx.x;
    if (i >= n) return;
    int g = (int)(i >> 7), c = (int)(i & 127);
    const float* gir = gi + (size_t)g * 384;
    const float* ghr = gh + (size_t)g * 384;
    float r  = sigf(gir[c] + ghr[c]);
    float z  = sigf(gir[128 + c] + ghr[128 + c]);
    float nn = tanhf(gir[256 + c] + r * ghr[256 + c]);
    h[i] = (1.f - z) * nn + z * h[i];
}

__global__ void y_gather_k(const float* __restrict__ h, float* __restrict__ y, long n)
{
    long i = (long)blockIdx.x * blockDim.x + threadIdx.x;
    if (i >= n) return;
    int b = (int)(i >> 7), c = (int)(i & 127);
    y[i] = h[(size_t)(b * 16 + 15) * EMB_ + c];
}

__global__ void build_M_k(const float* __restrict__ h, float* __restrict__ mb, long n)
{
    long i = (long)blockIdx.x * blockDim.x + threadIdx.x;
    if (i >= n) return;
    int g = (int)(i >> 8), c = (int)(i & 255);
    mb[i] = (c < 128) ? h[(size_t)g * 128 + c]
                      : h[(size_t)((g >> 4) * 16 + 15) * 128 + (c - 128)];
}

// ------------------------- estimator heads ---------------------------------
__global__ void est_heads_k(const float* __restrict__ z3,
    const float* We0, const float* be0, const float* We1, const float* be1,
    const float* We2, const float* be2,
    const float* Wv0, const float* bv0, const float* Wv1, const float* bv1,
    const float* Wv2, const float* bv2,
    const float* Wf0, const float* bf0, const float* Wf1, const float* bf1,
    float* __restrict__ out)
{
    int b = blockIdx.x * blockDim.x + threadIdx.x;
    if (b >= BB_) return;
    const float* zr = z3 + (size_t)b * 64;
    float t1[32], t2[16];
    // e head: 64->32->16->1 sigmoid
    for (int o = 0; o < 32; ++o) {
        float s = be0[o];
        for (int k = 0; k < 64; ++k) s += zr[k] * We0[k * 32 + o];
        t1[o] = fmaxf(s, 0.f);
    }
    for (int o = 0; o < 16; ++o) {
        float s = be1[o];
        for (int k = 0; k < 32; ++k) s += t1[k] * We1[k * 16 + o];
        t2[o] = fmaxf(s, 0.f);
    }
    float se = be2[0];
    for (int k = 0; k < 16; ++k) se += t2[k] * We2[k];
    float xe = sigf(se);
    // v head: 64->32->16->1 softplus
    for (int o = 0; o < 32; ++o) {
        float s = bv0[o];
        for (int k = 0; k < 64; ++k) s += zr[k] * Wv0[k * 32 + o];
        t1[o] = fmaxf(s, 0.f);
    }
    for (int o = 0; o < 16; ++o) {
        float s = bv1[o];
        for (int k = 0; k < 32; ++k) s += t1[k] * Wv1[k * 16 + o];
        t2[o] = fmaxf(s, 0.f);
    }
    float sv = bv2[0];
    for (int k = 0; k < 16; ++k) sv += t2[k] * Wv2[k];
    float xv = (sv > 20.f) ? sv : log1pf(expf(sv));
    // fs: [xe,xv] @ (2,8) + b  @ (8,1) + b  -> sigmoid (no inner activation)
    float acc = bf1[0];
    for (int o = 0; o < 8; ++o) {
        float u = xe * Wf0[o] + xv * Wf0[8 + o] + bf0[o];
        acc += u * Wf1[o];
    }
    out[b]           = xe;
    out[BB_ + b]     = xv;
    out[2 * BB_ + b] = sigf(acc);
}

// explainer final layer: (2048 x 128) @ (128 x 1) -> sigmoid
__global__ void expl_out_k(const float* __restrict__ m2, const float* __restrict__ Wx,
                           const float* __restrict__ bx, float* __restrict__ out)
{
    long idx = (long)blockIdx.x * blockDim.x + threadIdx.x;
    int lane = threadIdx.x & 31;
    long g = idx >> 5;
    if (g >= GG_) return;
    const float* r = m2 + (size_t)g * 128;
    float s = 0.f;
#pragma unroll
    for (int c = 0; c < 4; ++c) s += r[lane + 32 * c] * Wx[lane + 32 * c];
#pragma unroll
    for (int o = 16; o > 0; o >>= 1) s += __shfl_xor(s, o, 32);
    if (lane == 0) out[3 * BB_ + g] = sigf(s + bx[0]);
}

// ---------------------------------------------------------------------------
extern "C" void kernel_launch(void* const* d_in, const int* in_sizes, int n_in,
                              void* d_out, int out_size, void* d_ws, size_t ws_size,
                              hipStream_t stream)
{
    (void)in_sizes; (void)n_in; (void)out_size; (void)ws_size;
    auto F = [&](int i) -> const float* { return (const float*)d_in[i]; };
    const float* nf   = F(0);
    const int*   srcA = (const int*)d_in[1];
    const int*   dstA = srcA + N_NODES;
    float* out = (float*)d_out;

    // --- workspace carve (256B aligned) ---
    char* base = (char*)d_ws;
    size_t off = 0;
    auto carve = [&](size_t bytes) -> char* {
        char* p = base + off;
        off += (bytes + 255) & ~(size_t)255;
        return p;
    };
    float* XF   = (float*)carve((size_t)N_NODES * NFDP * 4);
    float* Qb   = (float*)carve((size_t)N_NODES * 256 * 4);
    float* Kb   = (float*)carve((size_t)N_NODES * 256 * 4);
    float* Vb   = (float*)carve((size_t)N_NODES * 256 * 4);
    float* AGGI = (float*)carve((size_t)N_NODES * 128 * 4);
    float* AGGO = (float*)carve((size_t)N_NODES * 128 * 4);
    float* XLIN = (float*)carve((size_t)N_NODES * 128 * 4);
    float* LOGI = (float*)carve((size_t)N_NODES * 2 * 4);
    float* SMAX = (float*)carve((size_t)N_NODES * 2 * 4);
    float* SSUM = (float*)carve((size_t)N_NODES * 2 * 4);
    float* BNS  = (float*)carve(128 * 4);
    float* BNQ  = (float*)carve(128 * 4);
    float* BNMU = (float*)carve(128 * 4);
    float* BNIV = (float*)carve(128 * 4);
    __bf16* WT  = (__bf16*)carve((size_t)512 * 512 * 2);
    __bf16* WIH = (__bf16*)carve((size_t)384 * 128 * 2);
    __bf16* WHH = (__bf16*)carve((size_t)384 * 128 * 2);
    float* H    = (float*)carve((size_t)GG_ * EMB_ * 4);
    // phase-disjoint aliases
    float* XSEQ = Qb;                      // after GNN
    float* GI   = Kb;                      // 2048x384
    float* GH   = Kb + (size_t)GG_ * 384;
    float* Y    = Vb;                      // 128x128
    float* Z1   = Y + 128 * 128;           // 128x512
    float* Z2   = Z1 + 128 * 512;          // 128x128
    float* Z3   = Z2 + 128 * 128;          // 128x64
    float* MB   = AGGI;                    // 2048x256
    float* M1   = AGGO;                    // 2048x256
    float* M2   = XLIN;                    // 2048x128

    auto grid1 = [](long n, int t) { return dim3((unsigned)((n + t - 1) / t)); };
    auto fill  = [&](float* p, float v, long n) {
        fill_f32<<<grid1(n, 256), 256, 0, stream>>>(p, v, n);
    };
    auto prepw = [&](const float* W, int Kdim, int Nout, int Kpad) {
        long n = (long)Nout * Kpad;
        transpose_w_k<<<grid1(n, 256), 256, 0, stream>>>(W, WT, Kdim, Nout, Kpad, n);
    };
    auto gemm = [&](const float* A, int lda, const __bf16* Wt, int ldw,
                    const float* bias, float* C, int ldc,
                    int M, int Nout, int K, int relu) {
        long thr = (long)(M / 32) * (Nout / 32) * 32;   // one wave per 32x32 tile
        gemm_bf16_wmma<<<grid1(thr, 128), 128, 0, stream>>>(
            A, lda, Wt, ldw, bias, C, ldc, M, Nout, K, relu);
    };

    // ============ encoder ============
    prepw(F(2), NT_, NTE_, 32);                                       // Wnt
    gemm(nf, RAW, WT, 32, F(3), XF, NFDP, N_NODES, NTE_, 32, 1);      // nt_enc -> x[:,0:32]
    copy_stats_tbl<<<grid1((long)N_NODES * 22, 256), 256, 0, stream>>>(nf, XF, (long)N_NODES * 22);
    colenc_k<<<grid1((long)N_NODES * 320, 256), 256, 0, stream>>>(nf, F(4), XF, (long)N_NODES * 320);
    padx_k<<<grid1((long)N_NODES * 10, 256), 256, 0, stream>>>(XF, (long)N_NODES * 10);

    // ============ 3 GNN layers ============
    // params per layer l at base 5+16*l:
    //   +0..5  in  (Wq,bq,Wk,bk,Wv,bv), +6..11 out (same), +12 Wr, +13 br, +14 g, +15 b
    int Kpad = NFDP, Kdim = NFD_;
    long ethreads = (long)N_NODES * 2 * 32;
    for (int l = 0; l < 3; ++l) {
        int pb = 5 + l * 16;
        for (int dir = 0; dir < 2; ++dir) {
            int db = pb + dir * 6;
            prepw(F(db + 0), Kdim, 256, Kpad);
            gemm(XF, NFDP, WT, Kpad, F(db + 1), Qb, 256, N_NODES, 256, Kpad, 0);
            prepw(F(db + 2), Kdim, 256, Kpad);
            gemm(XF, NFDP, WT, Kpad, F(db + 3), Kb, 256, N_NODES, 256, Kpad, 0);
            prepw(F(db + 4), Kdim, 256, Kpad);
            gemm(XF, NFDP, WT, Kpad, F(db + 5), Vb, 256, N_NODES, 256, Kpad, 0);
            const int* sA = (dir == 0) ? srcA : dstA;  // 'in': (src,dst); 'out': swapped
            const int* dA = (dir == 0) ? dstA : srcA;
            float* AGG = (dir == 0) ? AGGI : AGGO;
            fill(SMAX, -1e30f, (long)N_NODES * 2);
            fill(SSUM, 0.f,    (long)N_NODES * 2);
            fill(AGG,  0.f,    (long)N_NODES * 128);
            edge_logits_k<<<grid1(ethreads, 256), 256, 0, stream>>>(Qb, Kb, sA, dA, LOGI, N_NODES);
            seg_max_k<<<grid1((long)N_NODES * 2, 256), 256, 0, stream>>>(LOGI, dA, SMAX, (long)N_NODES * 2);
            seg_expsum_k<<<grid1((long)N_NODES * 2, 256), 256, 0, stream>>>(LOGI, dA, SMAX, SSUM, (long)N_NODES * 2);
            edge_agg_k<<<grid1(ethreads, 256), 256, 0, stream>>>(LOGI, Vb, sA, dA, SSUM, AGG, N_NODES);
        }
        prepw(F(pb + 12), Kdim, 128, Kpad);
        gemm(XF, NFDP, WT, Kpad, F(pb + 13), XLIN, 128, N_NODES, 128, Kpad, 0);
        fill(BNS, 0.f, 128);
        fill(BNQ, 0.f, 128);
        combine_k<<<grid1((long)N_NODES * 128, 256), 256, 0, stream>>>(AGGI, AGGO, XLIN, BNS, BNQ, (long)N_NODES * 128);
        bn_final_k<<<1, 128, 0, stream>>>(BNS, BNQ, BNMU, BNIV);
        bn_apply_k<<<grid1((long)N_NODES * 128, 256), 256, 0, stream>>>(XLIN, BNMU, BNIV, F(pb + 14), F(pb + 15), XF, (long)N_NODES * 128);
        Kpad = 128; Kdim = 128;
    }

    // ============ GRU ============  (params: 53 Wih, 54 Whh, 55 bih, 56 bhh)
    gru_reorder_k<<<grid1((long)N_NODES * 128, 256), 256, 0, stream>>>(XF, XSEQ, (long)N_NODES * 128);
    convert_w_k<<<grid1((long)384 * 128, 256), 256, 0, stream>>>(F(53), WIH, (long)384 * 128);
    convert_w_k<<<grid1((long)384 * 128, 256), 256, 0, stream>>>(F(54), WHH, (long)384 * 128);
    fill(H, 0.f, (long)GG_ * EMB_);
    for (int t = 0; t < LL_; ++t) {
        gemm(XSEQ + (size_t)t * GG_ * EMB_, EMB_, WIH, 128, F(55), GI, 384, GG_, 384, 128, 0);
        gemm(H, EMB_, WHH, 128, F(56), GH, 384, GG_, 384, 128, 0);
        gru_update_k<<<grid1((long)GG_ * EMB_, 256), 256, 0, stream>>>(GI, GH, H, (long)GG_ * EMB_);
    }

    // ============ estimator ============ (fcn 57..62, e 63..68, v 69..74, fs 75..78)
    y_gather_k<<<grid1((long)BB_ * EMB_, 256), 256, 0, stream>>>(H, Y, (long)BB_ * EMB_);
    prepw(F(57), 128, 512, 128);
    gemm(Y, 128, WT, 128, F(58), Z1, 512, BB_, 512, 128, 1);
    prepw(F(59), 512, 128, 512);
    gemm(Z1, 512, WT, 512, F(60), Z2, 128, BB_, 128, 512, 1);
    prepw(F(61), 128, 64, 128);
    gemm(Z2, 128, WT, 128, F(62), Z3, 64, BB_, 64, 128, 1);
    est_heads_k<<<1, 128, 0, stream>>>(Z3,
        F(63), F(64), F(65), F(66), F(67), F(68),
        F(69), F(70), F(71), F(72), F(73), F(74),
        F(75), F(76), F(77), F(78), out);

    // ============ explainer ============ (exp 79..84)
    build_M_k<<<grid1((long)GG_ * 256, 256), 256, 0, stream>>>(H, MB, (long)GG_ * 256);
    prepw(F(79), 256, 256, 256);
    gemm(MB, 256, WT, 256, F(80), M1, 256, GG_, 256, 256, 1);
    prepw(F(81), 256, 128, 256);
    gemm(M1, 256, WT, 256, F(82), M2, 128, GG_, 128, 256, 1);
    expl_out_k<<<grid1((long)GG_ * 32, 256), 256, 0, stream>>>(M2, F(83), F(84), out);
}
